// mLSTMCell_52312701665695
// MI455X (gfx1250) — compile-verified, matrix operand order checked by reading, not run on hardware
//
#include <hip/hip_runtime.h>
#include <hip/hip_bf16.h>
#include <math.h>

// mLSTM cell step, B=64, I=H=1024.
// d_in: 0 input[64,1024] 1 prev_h 2 prev_c[64,1024,1024] 3 prev_n[64,1024]
//       4 if_weights[2,1024] 5 o_weights[1024,1024] 6 kvq_weights[3072,1024]
//       7 if_bias[2] 8 o_bias[1024] 9 kvq_bias[3,1024]
// d_out: h[64*1024] ++ c[64*1024*1024] ++ n[64*1024]

#define BN 64
#define HN 1024

typedef __attribute__((ext_vector_type(16))) _Float16 v16h;
typedef __attribute__((ext_vector_type(8)))  float    v8f;

// ---------- workspace layout (floats) ----------
#define WS_I     0
#define WS_F     64
#define WS_INRM  128
#define WS_DEN   192
#define WS_O     256
#define WS_K     (WS_O + BN*HN)
#define WS_V     (WS_K + BN*HN)
#define WS_Q     (WS_V + BN*HN)

// ---------------- kernel 0: per-batch norm + exp gates ----------------
__global__ void __launch_bounds__(256) gates_kernel(const float* __restrict__ x,
                                                    const float* __restrict__ ifw,
                                                    const float* __restrict__ ifb,
                                                    float* __restrict__ ws) {
  int b = blockIdx.x;
  const float* xb = x + b * HN;
  float s0 = 0.f, s1 = 0.f, s2 = 0.f;
  for (int t = threadIdx.x; t < HN; t += 256) {
    float v = xb[t];
    s0 += v * v;
    s1 += v * ifw[t];
    s2 += v * ifw[HN + t];
  }
  __shared__ float r0[256], r1[256], r2[256];
  r0[threadIdx.x] = s0; r1[threadIdx.x] = s1; r2[threadIdx.x] = s2;
  __syncthreads();
  for (int st = 128; st > 0; st >>= 1) {
    if (threadIdx.x < st) {
      r0[threadIdx.x] += r0[threadIdx.x + st];
      r1[threadIdx.x] += r1[threadIdx.x + st];
      r2[threadIdx.x] += r2[threadIdx.x + st];
    }
    __syncthreads();
  }
  if (threadIdx.x == 0) {
    ws[WS_INRM + b] = rsqrtf(r0[0]);              // 1/||x||
    ws[WS_I + b]    = expf(r1[0] + ifb[0]);       // input gate
    ws[WS_F + b]    = expf(r2[0] + ifb[1]);       // forget gate
  }
}

// ---------------- WMMA fragment load: f32 row -> v16h fragment ----------
// ISA 16-bit A layout (and symmetric B layout on row-major W):
// lane l: row = base + (l&15); half=(l>>4); K elems = {kk+half*8 .. +7, kk+half*8+16 .. +7}
__device__ __forceinline__ v16h load_frag(const float* __restrict__ rowp, int kk, int half) {
  const float4* p0 = (const float4*)(rowp + kk + half * 8);
  const float4* p1 = (const float4*)(rowp + kk + half * 8 + 16);
  float4 a0 = p0[0], a1 = p0[1], b0 = p1[0], b1 = p1[1];
  v16h r;
  r[0]  = (_Float16)a0.x; r[1]  = (_Float16)a0.y;
  r[2]  = (_Float16)a0.z; r[3]  = (_Float16)a0.w;
  r[4]  = (_Float16)a1.x; r[5]  = (_Float16)a1.y;
  r[6]  = (_Float16)a1.z; r[7]  = (_Float16)a1.w;
  r[8]  = (_Float16)b0.x; r[9]  = (_Float16)b0.y;
  r[10] = (_Float16)b0.z; r[11] = (_Float16)b0.w;
  r[12] = (_Float16)b1.x; r[13] = (_Float16)b1.y;
  r[14] = (_Float16)b1.z; r[15] = (_Float16)b1.w;
  return r;
}

// ------- kernel 1: Y = X @ [o_w;kvq_w]^T via WMMA, fused o/k/v/q/n epilogue ---
// One wave per 16x16 output tile. M tiles = 4 (batch), N tiles = 256 (4096 cols).
__global__ void __launch_bounds__(128) proj_wmma_kernel(
    const float* __restrict__ x, const float* __restrict__ ow,
    const float* __restrict__ kvqw, const float* __restrict__ ob,
    const float* __restrict__ kvqb, const float* __restrict__ prev_n,
    const float* __restrict__ wsc, float* __restrict__ o_out,
    float* __restrict__ k_out, float* __restrict__ v_out,
    float* __restrict__ q_out, float* __restrict__ n_out) {
  int wave = threadIdx.x >> 5;
  int lane = threadIdx.x & 31;
  int tile = blockIdx.x * 4 + wave;       // 0..1023
  int mt = tile & 3;                      // batch tile
  int nt = tile >> 2;                     // column tile (uniform per wave)
  int m0 = mt * 16, n0 = nt * 16;
  int half = lane >> 4, r = lane & 15;

  const float* xrow = x + (size_t)(m0 + r) * HN;
  const float* wrow = (n0 < HN) ? (ow + (size_t)(n0 + r) * HN)
                                : (kvqw + (size_t)(n0 - HN + r) * HN);
  v8f acc = {};
#pragma unroll 2
  for (int kk = 0; kk < HN; kk += 32) {
    v16h af = load_frag(xrow, kk, half);
    v16h bf = load_frag(wrow, kk, half);
    acc = __builtin_amdgcn_wmma_f32_16x16x32_f16(
        /*neg_a=*/false, af, /*neg_b=*/false, bf,
        /*c_mod=*/(short)0, acc, /*reuse_a=*/false, /*reuse_b=*/false);
  }

  // Epilogue: branch on wave-uniform n0 (tiles never straddle a 1024 boundary),
  // fully unrolled so acc[rr] is a direct VGPR reference (no movrel).
  int colr = n0 + r;
  if (n0 < HN) {                           // ---- output gate ----
    float obias = ob[colr];
#pragma unroll
    for (int rr = 0; rr < 8; ++rr) {
      int b = m0 + rr + 8 * half;
      o_out[b * HN + colr] = 1.f / (1.f + expf(-(acc[rr] + obias)));
    }
  } else if (n0 < 2 * HN) {                // ---- k path (+ n state) ----
    int feat = colr - HN;
    float kb = kvqb[feat];
#pragma unroll
    for (int rr = 0; rr < 8; ++rr) {
      int b = m0 + rr + 8 * half;
      float kx = acc[rr] * wsc[WS_INRM + b] * 0.03125f + kb;   // 1/sqrt(1024)
      k_out[b * HN + feat] = kx;
      n_out[b * HN + feat] = wsc[WS_F + b] * prev_n[b * HN + feat]
                           + wsc[WS_I + b] * kx;
    }
  } else if (n0 < 3 * HN) {                // ---- v path ----
    int feat = colr - 2 * HN;
    float vb = kvqb[HN + feat];
#pragma unroll
    for (int rr = 0; rr < 8; ++rr) {
      int b = m0 + rr + 8 * half;
      v_out[b * HN + feat] = acc[rr] * wsc[WS_INRM + b] + vb;
    }
  } else {                                 // ---- q path ----
    int feat = colr - 3 * HN;
    float qb = kvqb[2 * HN + feat];
#pragma unroll
    for (int rr = 0; rr < 8; ++rr) {
      int b = m0 + rr + 8 * half;
      q_out[b * HN + feat] = acc[rr] * wsc[WS_INRM + b] + qb;
    }
  }
}

// ---------------- kernel 2: denom[b] = max(|n . q|, 1) ----------------
__global__ void __launch_bounds__(256) denom_kernel(const float* __restrict__ n_st,
                                                    const float* __restrict__ q,
                                                    float* __restrict__ ws) {
  int b = blockIdx.x;
  float s = 0.f;
  for (int t = threadIdx.x; t < HN; t += 256)
    s += n_st[b * HN + t] * q[b * HN + t];
  __shared__ float red[256];
  red[threadIdx.x] = s;
  __syncthreads();
  for (int st = 128; st > 0; st >>= 1) {
    if (threadIdx.x < st) red[threadIdx.x] += red[threadIdx.x + st];
    __syncthreads();
  }
  if (threadIdx.x == 0) ws[WS_DEN + b] = fmaxf(fabsf(red[0]), 1.0f);
}

// ------- kernel 3: stream prev_c -> c, fused c@q dot, h readout ---------
// One wave32 per (b,row). 8 waves/block, block-uniform batch; k,q staged in LDS.
__global__ void __launch_bounds__(256) cell_update_kernel(
    const float* __restrict__ prev_c, const float* __restrict__ wsc,
    const float* __restrict__ o_v, const float* __restrict__ k_v,
    const float* __restrict__ v_v, const float* __restrict__ q_v,
    float* __restrict__ h_out, float* __restrict__ c_out) {
  __shared__ float ks[HN], qs[HN];
  int wid  = threadIdx.x >> 5;
  int lane = threadIdx.x & 31;
  int wbase = blockIdx.x * 8;             // first row-id of block
  int b = wbase >> 10;                    // uniform: 1024 rows per batch, 8 | 1024

  int t4 = threadIdx.x * 4;               // stage k,q (4 KB each) into LDS
  *(float4*)(ks + t4) = *(const float4*)(k_v + b * HN + t4);
  *(float4*)(qs + t4) = *(const float4*)(q_v + b * HN + t4);
  __syncthreads();

  int row = (wbase & (HN - 1)) + wid;
  float fg = wsc[WS_F + b];
  float iv = wsc[WS_I + b] * v_v[b * HN + row];
  float og = o_v[b * HN + row];
  float dn = wsc[WS_DEN + b];

  const float* pc = prev_c + (size_t)(b * HN + row) * HN;
  float*       cc = c_out  + (size_t)(b * HN + row) * HN;

  float acc = 0.f;
#pragma unroll
  for (int it = 0; it < 8; ++it) {
    int col = it * 128 + lane * 4;
    float4 p  = *(const float4*)(pc + col);
    float4 k4 = *(const float4*)(ks + col);
    float4 q4 = *(const float4*)(qs + col);
    float4 cv;
    cv.x = fg * p.x + iv * k4.x;
    cv.y = fg * p.y + iv * k4.y;
    cv.z = fg * p.z + iv * k4.z;
    cv.w = fg * p.w + iv * k4.w;
    *(float4*)(cc + col) = cv;
    acc += cv.x * q4.x + cv.y * q4.y + cv.z * q4.z + cv.w * q4.w;
  }
  for (int off = 16; off > 0; off >>= 1) acc += __shfl_xor(acc, off, 32);
  if (lane == 0) h_out[b * HN + row] = og * (acc / dn);
}

// ------------------------------- launcher -------------------------------
extern "C" void kernel_launch(void* const* d_in, const int* in_sizes, int n_in,
                              void* d_out, int out_size, void* d_ws, size_t ws_size,
                              hipStream_t stream) {
  const float* input   = (const float*)d_in[0];
  const float* prev_c  = (const float*)d_in[2];
  const float* prev_n  = (const float*)d_in[3];
  const float* if_w    = (const float*)d_in[4];
  const float* o_w     = (const float*)d_in[5];
  const float* kvq_w   = (const float*)d_in[6];
  const float* if_b    = (const float*)d_in[7];
  const float* o_b     = (const float*)d_in[8];
  const float* kvq_b   = (const float*)d_in[9];

  float* out   = (float*)d_out;
  float* h_out = out;                                    // [64*1024]
  float* c_out = out + BN * HN;                          // [64*1024*1024]
  float* n_out = out + BN * HN + (size_t)BN * HN * HN;   // [64*1024]

  float* ws = (float*)d_ws;
  float* o_s = ws + WS_O;
  float* k_s = ws + WS_K;
  float* v_s = ws + WS_V;
  float* q_s = ws + WS_Q;

  gates_kernel<<<BN, 256, 0, stream>>>(input, if_w, if_b, ws);
  proj_wmma_kernel<<<256, 128, 0, stream>>>(input, o_w, kvq_w, o_b, kvq_b,
                                            prev_n, ws, o_s, k_s, v_s, q_s, n_out);
  denom_kernel<<<BN, 256, 0, stream>>>(n_out, q_s, ws);
  cell_update_kernel<<<(BN * HN) / 8, 256, 0, stream>>>(prev_c, ws, o_s, k_s,
                                                        v_s, q_s, h_out, c_out);
}
